// QwenAttention_20495583936675
// MI455X (gfx1250) — matrix operand order repaired and checked by reading, not compiled, not run on hardware
//
#include <hip/hip_runtime.h>

// ---------------------------------------------------------------------------
// Problem constants (match the reference)
// ---------------------------------------------------------------------------
#define BATCH   2
#define SEQ     2048
#define HIDDEN  2048
#define NHEADS  16
#define HDIM    128           // head dim
#define BS      (BATCH*SEQ)   // 4096 rows

typedef __attribute__((ext_vector_type(16))) __bf16        bf16x16;
typedef __attribute__((ext_vector_type(8)))  float         f32x8;
typedef __attribute__((ext_vector_type(4)))  float         f32x4;
typedef __attribute__((ext_vector_type(4)))  unsigned int  u32x4;
typedef __attribute__((ext_vector_type(2)))  unsigned int  u32x2;

// fp32 -> bf16, round-to-nearest-even
__device__ __forceinline__ unsigned short f2bf(float f) {
  unsigned int u = __float_as_uint(f);
  u += 0x7fffu + ((u >> 16) & 1u);
  return (unsigned short)(u >> 16);
}

__device__ __forceinline__ f32x8 wmma_bf16(bf16x16 a, bf16x16 b, f32x8 c) {
  // D = A(16x32 bf16) * B(32x16 bf16) + C(16x16 f32)
  return __builtin_amdgcn_wmma_f32_16x16x32_bf16(false, a, false, b,
                                                 (short)0, c, false, false);
}

// A-fragment (16x32 bf16) from a row-major LDS tile [row][k], stride ld (bf16 units).
// Lanes 0-15 (row M=lane) hold K={0..7,16..23}; lanes 16-31 hold K={8..15,24..31}.
__device__ __forceinline__ bf16x16 ldsA_frag(const unsigned short* lds,
                                             int row0, int ld, int kOff) {
  const int lane = threadIdx.x & 31;
  const int r = lane & 15, h = lane >> 4;
  const unsigned int* q =
      (const unsigned int*)(lds + (size_t)(row0 + r) * ld + kOff + h * 8);
  union { unsigned int u[8]; bf16x16 v; } x;
  x.u[0] = q[0];  x.u[1] = q[1];  x.u[2] = q[2];  x.u[3] = q[3];
  x.u[4] = q[8];  x.u[5] = q[9];  x.u[6] = q[10]; x.u[7] = q[11];
  return x.v;
}

// B-fragment (32x16 bf16) from LDS stored [col][k], stride ld.
// Lanes 0-15 (col N=lane) hold K={0..7,16..23}; lanes 16-31 hold K={8..15,24..31}.
__device__ __forceinline__ bf16x16 ldsB_frag(const unsigned short* lds,
                                             int col0, int ld, int kOff) {
  const int lane = threadIdx.x & 31;
  const int r = lane & 15, h = lane >> 4;
  const unsigned int* q =
      (const unsigned int*)(lds + (size_t)(col0 + r) * ld + kOff + h * 8);
  union { unsigned int u[8]; bf16x16 v; } x;
#pragma unroll
  for (int i = 0; i < 8; ++i) x.u[i] = q[(i & 3) + (i >> 2) * 8];
  return x.v;
}

// ---------------------------------------------------------------------------
// One-time precision passes
// ---------------------------------------------------------------------------
// fp32 -> bf16 elementwise (vectorized x4)
__global__ __launch_bounds__(256)
void convert_kernel(const float* __restrict__ in, unsigned short* __restrict__ out,
                    int n) {
  const int i = (blockIdx.x * 256 + threadIdx.x) * 4;
  if (i < n) {
    f32x4 v = *(const f32x4*)(in + i);
    unsigned int lo = (unsigned int)f2bf(v.x) | ((unsigned int)f2bf(v.y) << 16);
    unsigned int hi = (unsigned int)f2bf(v.z) | ((unsigned int)f2bf(v.w) << 16);
    u32x2 p = {lo, hi};
    *(u32x2*)(out + i) = p;
  }
}

// fp32 [K][N] -> bf16 [N][K] (tiled transpose through LDS)
__global__ __launch_bounds__(256)
void transpose_convert_kernel(const float* __restrict__ in,
                              unsigned short* __restrict__ out, int K, int N) {
  __shared__ unsigned short t[32 * 33];
  const int tn = blockIdx.x, tk = blockIdx.y;
  const int tid = threadIdx.x;
  {
    const int r  = tid >> 3;        // k within tile, 0..31
    const int c4 = (tid & 7) * 4;   // n within tile
    f32x4 v = *(const f32x4*)(in + (size_t)(tk * 32 + r) * N + tn * 32 + c4);
    t[(c4 + 0) * 33 + r] = f2bf(v.x);
    t[(c4 + 1) * 33 + r] = f2bf(v.y);
    t[(c4 + 2) * 33 + r] = f2bf(v.z);
    t[(c4 + 3) * 33 + r] = f2bf(v.w);
  }
  __syncthreads();
  {
    const int nr = tid >> 3;        // n within tile
    const int k4 = (tid & 7) * 4;   // k within tile
    unsigned short* o = out + (size_t)(tn * 32 + nr) * K + tk * 32 + k4;
    o[0] = t[nr * 33 + k4 + 0];
    o[1] = t[nr * 33 + k4 + 1];
    o[2] = t[nr * 33 + k4 + 2];
    o[3] = t[nr * 33 + k4 + 3];
  }
}

// ---------------------------------------------------------------------------
// GEMM: C[M,N] = A[M,K] @ B[K,N], A bf16 row-major, Bt bf16 [n][k] (pre-transposed).
// Block tile 128x128, BK=64, 8 waves (4Mx2N), wave = 32x64 via 2x4 accumulators.
// Double-buffered LDS: global loads for tile i+1 overlap WMMA of tile i,
// one barrier per K-step.
// MODE 0: bf16 out [m][n];  MODE 1: bf16 out transposed [n][m];  MODE 2: f32 out.
// ---------------------------------------------------------------------------
#define GLD 80   // LDS row stride (bf16): 160B, 16B-aligned fragments

template <int MODE>
__global__ __launch_bounds__(256)
void gemm_bf16_kernel(const unsigned short* __restrict__ A,
                      const unsigned short* __restrict__ Bt,
                      void* __restrict__ Cout, int M, int N, int K) {
  __shared__ unsigned short sA[2][128 * GLD];   // [m][k] ping-pong
  __shared__ unsigned short sB[2][128 * GLD];   // [n][k] ping-pong

  const int tile_n = blockIdx.x, tile_m = blockIdx.y;
  const int tid  = threadIdx.x;
  const int wave = tid >> 5;
  const int wm   = wave & 3;   // m-offset = wm*32
  const int wn   = wave >> 2;  // n-offset = wn*64
  const int lane = tid & 31;
  const int lr   = lane & 15, lh = lane >> 4;

  // staging coordinates: 128 rows x 64 bf16, 256 threads, 16B copies
  const int c8 = (tid & 7) * 8;   // 0..56
  const int r0 = tid >> 3;        // 0..31
  const unsigned short* gA = A  + (size_t)(tile_m * 128 + r0) * K + c8;
  const unsigned short* gB = Bt + (size_t)(tile_n * 128 + r0) * K + c8;
  const size_t rstride = (size_t)32 * K;   // 32 rows forward

  f32x8 acc[2][4];
#pragma unroll
  for (int i = 0; i < 2; ++i)
#pragma unroll
    for (int j = 0; j < 4; ++j) acc[i][j] = (f32x8)(0.0f);

  // prologue: fetch tile 0 into registers
  u32x4 ra[4], rb[4];
#pragma unroll
  for (int rr = 0; rr < 4; ++rr) {
    ra[rr] = *(const u32x4*)(gA + rr * rstride);
    rb[rr] = *(const u32x4*)(gB + rr * rstride);
  }

  int buf = 0;
  for (int k0 = 0; k0 < K; k0 += 64) {
    // commit staged registers to this iteration's LDS buffer
#pragma unroll
    for (int rr = 0; rr < 4; ++rr) {
      const int row = r0 + rr * 32;
      *(u32x4*)(sA[buf] + (size_t)row * GLD + c8) = ra[rr];
      *(u32x4*)(sB[buf] + (size_t)row * GLD + c8) = rb[rr];
    }
    __syncthreads();
    // issue next tile's global loads; they retire under the WMMAs below
    if (k0 + 64 < K) {
#pragma unroll
      for (int rr = 0; rr < 4; ++rr) {
        ra[rr] = *(const u32x4*)(gA + (k0 + 64) + rr * rstride);
        rb[rr] = *(const u32x4*)(gB + (k0 + 64) + rr * rstride);
      }
    }
#pragma unroll
    for (int kc = 0; kc < 2; ++kc) {
      bf16x16 af0 = ldsA_frag(sA[buf], wm * 32,      GLD, kc * 32);
      bf16x16 af1 = ldsA_frag(sA[buf], wm * 32 + 16, GLD, kc * 32);
#pragma unroll
      for (int nt = 0; nt < 4; ++nt) {
        bf16x16 bf = ldsB_frag(sB[buf], wn * 64 + nt * 16, GLD, kc * 32);
        acc[0][nt] = wmma_bf16(af0, bf, acc[0][nt]);
        acc[1][nt] = wmma_bf16(af1, bf, acc[1][nt]);
      }
    }
    buf ^= 1;   // next iteration writes the other buffer: one barrier per step
  }

  // ---- store (C/D layout: row = base + i + 8*(lane/16), col = base + lane%16)
  if constexpr (MODE == 2) {
    float* C = (float*)Cout;
#pragma unroll
    for (int mt = 0; mt < 2; ++mt)
#pragma unroll
      for (int nt = 0; nt < 4; ++nt)
#pragma unroll
        for (int i = 0; i < 8; ++i) {
          const int row = tile_m * 128 + wm * 32 + mt * 16 + i + 8 * lh;
          const int col = tile_n * 128 + wn * 64 + nt * 16 + lr;
          C[(size_t)row * N + col] = acc[mt][nt][i];
        }
  } else if constexpr (MODE == 0) {
    unsigned short* C = (unsigned short*)Cout;
#pragma unroll
    for (int mt = 0; mt < 2; ++mt)
#pragma unroll
      for (int nt = 0; nt < 4; ++nt)
#pragma unroll
        for (int i = 0; i < 8; ++i) {
          const int row = tile_m * 128 + wm * 32 + mt * 16 + i + 8 * lh;
          const int col = tile_n * 128 + wn * 64 + nt * 16 + lr;
          C[(size_t)row * N + col] = f2bf(acc[mt][nt][i]);
        }
  } else {  // MODE 1: transposed bf16 [n][m]; lane's 8 rows are contiguous in m
    unsigned short* C = (unsigned short*)Cout;
#pragma unroll
    for (int mt = 0; mt < 2; ++mt)
#pragma unroll
      for (int nt = 0; nt < 4; ++nt) {
        const int rowb = tile_m * 128 + wm * 32 + mt * 16 + 8 * lh;
        const int col  = tile_n * 128 + wn * 64 + nt * 16 + lr;
        unsigned int w0 = (unsigned int)f2bf(acc[mt][nt][0]) |
                          ((unsigned int)f2bf(acc[mt][nt][1]) << 16);
        unsigned int w1 = (unsigned int)f2bf(acc[mt][nt][2]) |
                          ((unsigned int)f2bf(acc[mt][nt][3]) << 16);
        unsigned int w2 = (unsigned int)f2bf(acc[mt][nt][4]) |
                          ((unsigned int)f2bf(acc[mt][nt][5]) << 16);
        unsigned int w3 = (unsigned int)f2bf(acc[mt][nt][6]) |
                          ((unsigned int)f2bf(acc[mt][nt][7]) << 16);
        u32x4 pk = {w0, w1, w2, w3};
        *(u32x4*)(C + (size_t)col * M + rowb) = pk;
      }
  }
}

// ---------------------------------------------------------------------------
// Flash attention (causal, online softmax). bf16 Q/K/V^T in, bf16 context out.
// One block per (128-row q-block, head, batch); 8 waves, each owns 16 rows.
// ---------------------------------------------------------------------------
#define LDT 136   // LDS row stride (bf16): 272B, conflict-friendly, 16B aligned

__global__ __launch_bounds__(256)
void flash_attn_kernel(const unsigned short* __restrict__ Q,
                       const unsigned short* __restrict__ Kb,
                       const unsigned short* __restrict__ Vt,  // [HIDDEN][BS]
                       unsigned short* __restrict__ Ctx) {
  __shared__ unsigned short sK[128 * LDT];   // Q tile first, then K tiles [k][d]
  __shared__ unsigned short sVt[128 * LDT];  // V^T tile [d][k]
  __shared__ unsigned short sP[128 * LDT];   // per-wave 16-row P slabs [m][k]

  const int qb = blockIdx.x, head = blockIdx.y, b = blockIdx.z;
  const int tid  = threadIdx.x;
  const int wave = tid >> 5;
  const int lane = tid & 31;
  const int lr   = lane & 15, lh = lane >> 4;
  const float sm_scale = 0.08838834764831845f;   // 1/sqrt(128)

  // staging coords: 128x128 bf16 tile, 256 threads, 16B copies
  const int c8 = (tid & 15) * 8;  // 0..120
  const int r0 = tid >> 4;        // 0..15

  // ---- stage Q tile into sK, pull persistent A-fragments ----
  {
    const size_t base =
        (size_t)(b * SEQ + qb * 128) * HIDDEN + head * HDIM;
#pragma unroll
    for (int rr = 0; rr < 128; rr += 16) {
      const int row = r0 + rr;
      *(u32x4*)(sK + (size_t)row * LDT + c8) =
          *(const u32x4*)(Q + base + (size_t)row * HIDDEN + c8);
    }
  }
  __syncthreads();
  bf16x16 qf[4];
#pragma unroll
  for (int kc = 0; kc < 4; ++kc) qf[kc] = ldsA_frag(sK, wave * 16, LDT, kc * 32);

  f32x8 o[8];
#pragma unroll
  for (int dt = 0; dt < 8; ++dt) o[dt] = (f32x8)(0.0f);
  float mi[8], li[8];
#pragma unroll
  for (int i = 0; i < 8; ++i) { mi[i] = -1e30f; li[i] = 0.0f; }

  const int nkb = qb + 1;   // causal: only key blocks <= query block
  for (int kb = 0; kb < nkb; ++kb) {
    __syncthreads();   // prior iteration's reads done before restage
    // ---- stage K [k][d] and V^T [d][k] (both straight 16B row copies) ----
    {
      const size_t baseK =
          (size_t)(b * SEQ + kb * 128) * HIDDEN + head * HDIM;
      const size_t baseV =
          (size_t)(head * HDIM) * BS + b * SEQ + kb * 128;
#pragma unroll
      for (int rr = 0; rr < 128; rr += 16) {
        const int row = r0 + rr;
        *(u32x4*)(sK + (size_t)row * LDT + c8) =
            *(const u32x4*)(Kb + baseK + (size_t)row * HIDDEN + c8);
        *(u32x4*)(sVt + (size_t)row * LDT + c8) =
            *(const u32x4*)(Vt + baseV + (size_t)row * BS + c8);
      }
    }
    __syncthreads();

    // ---- scores S = Q K^T (wave's 16 rows x 128 keys) ----
    f32x8 s[8];
#pragma unroll
    for (int nt = 0; nt < 8; ++nt) {
      s[nt] = (f32x8)(0.0f);
#pragma unroll
      for (int kc = 0; kc < 4; ++kc) {
        bf16x16 bf = ldsB_frag(sK, nt * 16, LDT, kc * 32);
        s[nt] = wmma_bf16(qf[kc], bf, s[nt]);
      }
    }
    // scale + causal mask (diagonal block only)
    const bool diag = (kb == qb);
#pragma unroll
    for (int nt = 0; nt < 8; ++nt)
#pragma unroll
      for (int i = 0; i < 8; ++i) {
        float v = s[nt][i] * sm_scale;
        if (diag) {
          const int qr  = wave * 16 + i + 8 * lh;
          const int kcl = nt * 16 + lr;
          if (kcl > qr) v = -1e30f;
        }
        s[nt][i] = v;
      }
    // ---- online softmax: per-row 16-lane shuffle reductions ----
    float alpha[8];
#pragma unroll
    for (int i = 0; i < 8; ++i) {
      float p = s[0][i];
#pragma unroll
      for (int nt = 1; nt < 8; ++nt) p = fmaxf(p, s[nt][i]);
#pragma unroll
      for (int off = 8; off > 0; off >>= 1) p = fmaxf(p, __shfl_xor(p, off, 16));
      const float nm = fmaxf(mi[i], p);
      alpha[i] = __expf(mi[i] - nm);
      mi[i] = nm;
    }
#pragma unroll
    for (int i = 0; i < 8; ++i) {
      float rs = 0.0f;
      const int prow = wave * 16 + i + 8 * lh;
#pragma unroll
      for (int nt = 0; nt < 8; ++nt) {
        const float e = __expf(s[nt][i] - mi[i]);
        rs += e;
        sP[(size_t)prow * LDT + nt * 16 + lr] = f2bf(e);
      }
#pragma unroll
      for (int off = 8; off > 0; off >>= 1) rs += __shfl_xor(rs, off, 16);
      li[i] = li[i] * alpha[i] + rs;
    }
#pragma unroll
    for (int dt = 0; dt < 8; ++dt)
#pragma unroll
      for (int i = 0; i < 8; ++i) o[dt][i] *= alpha[i];

    // ---- O += P V (wave reads back only its own sP slab) ----
    bf16x16 pf[4];
#pragma unroll
    for (int kc = 0; kc < 4; ++kc) pf[kc] = ldsA_frag(sP, wave * 16, LDT, kc * 32);
#pragma unroll
    for (int dt = 0; dt < 8; ++dt)
#pragma unroll
      for (int kc = 0; kc < 4; ++kc) {
        bf16x16 bf = ldsB_frag(sVt, dt * 16, LDT, kc * 32);
        o[dt] = wmma_bf16(pf[kc], bf, o[dt]);
      }
  }

  // ---- normalize and write bf16 context [B*S, HIDDEN] ----
  float inv[8];
#pragma unroll
  for (int i = 0; i < 8; ++i) inv[i] = 1.0f / li[i];
#pragma unroll
  for (int dt = 0; dt < 8; ++dt)
#pragma unroll
    for (int i = 0; i < 8; ++i) {
      const int row = b * SEQ + qb * 128 + wave * 16 + i + 8 * lh;
      const int col = head * HDIM + dt * 16 + lr;
      Ctx[(size_t)row * HIDDEN + col] = f2bf(o[dt][i] * inv[i]);
    }
}

// ---------------------------------------------------------------------------
// Launch: convert -> Q/K/V projections -> flash attention -> output projection
// ---------------------------------------------------------------------------
extern "C" void kernel_launch(void* const* d_in, const int* in_sizes, int n_in,
                              void* d_out, int out_size, void* d_ws, size_t ws_size,
                              hipStream_t stream) {
  (void)in_sizes; (void)n_in; (void)out_size; (void)ws_size;
  const float* hidden = (const float*)d_in[0];
  // d_in[1] = attention_mask (pure causal; applied analytically in-kernel)
  const float* wq = (const float*)d_in[2];
  const float* wk = (const float*)d_in[3];
  const float* wv = (const float*)d_in[4];
  const float* wo = (const float*)d_in[5];
  float* out = (float*)d_out;

  const size_t HE = (size_t)BS * HIDDEN;        // 8,388,608 elements
  const size_t WE = (size_t)HIDDEN * HIDDEN;    // 4,194,304 elements
  unsigned short* hB   = (unsigned short*)d_ws;
  unsigned short* wqT  = hB  + HE;
  unsigned short* wkT  = wqT + WE;
  unsigned short* wvT  = wkT + WE;
  unsigned short* woT  = wvT + WE;
  unsigned short* qB   = woT + WE;
  unsigned short* kB   = qB  + HE;
  unsigned short* vtB  = kB  + HE;   // V^T: [HIDDEN][BS]
  unsigned short* ctxB = vtB + HE;

  // one-time precision passes
  convert_kernel<<<(unsigned)(HE / 1024), 256, 0, stream>>>(hidden, hB, (int)HE);
  dim3 gt(HIDDEN / 32, HIDDEN / 32);
  transpose_convert_kernel<<<gt, 256, 0, stream>>>(wq, wqT, HIDDEN, HIDDEN);
  transpose_convert_kernel<<<gt, 256, 0, stream>>>(wk, wkT, HIDDEN, HIDDEN);
  transpose_convert_kernel<<<gt, 256, 0, stream>>>(wv, wvT, HIDDEN, HIDDEN);
  transpose_convert_kernel<<<gt, 256, 0, stream>>>(wo, woT, HIDDEN, HIDDEN);

  dim3 gg(HIDDEN / 128, BS / 128);   // (16, 32)
  gemm_bf16_kernel<0><<<gg, 256, 0, stream>>>(hB, wqT, qB,  BS, HIDDEN, HIDDEN);
  gemm_bf16_kernel<0><<<gg, 256, 0, stream>>>(hB, wkT, kB,  BS, HIDDEN, HIDDEN);
  gemm_bf16_kernel<1><<<gg, 256, 0, stream>>>(hB, wvT, vtB, BS, HIDDEN, HIDDEN);

  dim3 ga(SEQ / 128, NHEADS, BATCH);  // (16, 16, 2)
  flash_attn_kernel<<<ga, 256, 0, stream>>>(qB, kB, vtB, ctxB);

  gemm_bf16_kernel<2><<<gg, 256, 0, stream>>>(ctxB, woT, out, BS, HIDDEN, HIDDEN);
}
